// ParticleNetPyG_5196910428872
// MI455X (gfx1250) — matrix operand order repaired and checked by reading, not compiled, not run on hardware
//
#include <hip/hip_runtime.h>

// ---------------------------------------------------------------------------
// ParticleNet forward for MI455X (gfx1250, wave32, WMMA).
// All matmuls (edge MLPs, shortcuts, fusion, fc1) use v_wmma_f32_16x16x32_f16
// with f32 accumulation; BN+bias folded into per-channel alpha/beta epilogue.
// GEMM uses 4x2 register super-tiles per wave: 8 WMMAs per 12 b128 loads.
// ---------------------------------------------------------------------------

typedef __attribute__((ext_vector_type(16))) _Float16 v16h;
typedef __attribute__((ext_vector_type(8)))  _Float16 v8h;
typedef __attribute__((ext_vector_type(8)))  float    v8f;

union AFrag { v16h v; v8h h[2]; };

#define NPTS   16384      // B*P
#define PJET   128
#define KNN    16
#define CBJ    16         // jets per edge chunk
#define CHPTS  (CBJ*PJET)         // 2048 points per chunk
#define CHEDGE (CHPTS*KNN)        // 32768 edge rows per chunk

static __device__ __forceinline__ v8f wmma_f16(v16h a, v16h b, v8f c)
{
    return __builtin_amdgcn_wmma_f32_16x16x32_f16(false, a, false, b,
                                                  (short)0, c, false, false);
}

// ---------------------------------------------------------------------------
// Generic WMMA GEMM: out = act(A[M,K](f16,row-major,lda) x Bp(packed) * alpha[n] + beta[n])
// Bp layout: tile(kt,nt): 32 lanes x 16 halves contiguous;
//   Bp[((kt*NT+nt)*32 + lane)*16 + i] = W[kt*32 + (i<8 ? 0 : 16) + (lane>>4)*8 + (i&7)][nt*16 + (lane&15)]
// Each wave computes a 64x32 super-tile (4 M-tiles x 2 N-tiles, 8 accumulators).
// Requires M % 64 == 0, N % 32 == 0, K % 32 == 0.
// ---------------------------------------------------------------------------
__global__ __launch_bounds__(256) void gemm_wmma_kernel(
    const _Float16* __restrict__ A, int lda,
    const _Float16* __restrict__ Bp,
    _Float16* __restrict__ out16, float* __restrict__ out32, int ldo,
    const float* __restrict__ alpha, const float* __restrict__ beta,
    int M, int N, int K, int relu)
{
    const int lane   = threadIdx.x & 31;
    const int wave   = threadIdx.x >> 5;
    const int Ntiles = N >> 4;
    const int NT2    = N >> 5;             // super-tile columns
    const int MT4    = M >> 6;             // super-tile rows
    const int stiles = MT4 * NT2;
    int st = blockIdx.x * 8 + wave;
    if (st >= stiles) return;              // wave-uniform exit; EXEC stays all-1
    const int smt  = st / NT2;
    const int snt  = st - smt * NT2;
    const int lr   = lane & 15;
    const int half = lane >> 4;

    // A: 4 M-tiles, rows smt*64 + mi*16 + lr
    const _Float16* pa = A + (size_t)(smt * 64 + lr) * lda + half * 8;
    const size_t aRowStep = (size_t)16 * lda;        // next M-tile
    // B: 2 N-tiles starting at snt*2
    const _Float16* pb = Bp + ((size_t)(snt * 2) * 512 + (size_t)lane * 16);
    const size_t bStep = (size_t)Ntiles * 512;       // next kt

    v8f acc[8];
#pragma unroll
    for (int i = 0; i < 8; ++i) acc[i] = (v8f){};

    const int KT = K >> 5;
    for (int kt = 0; kt < KT; ++kt) {
        AFrag a[4];
#pragma unroll
        for (int mi = 0; mi < 4; ++mi) {
            const _Float16* p = pa + mi * aRowStep;
            a[mi].h[0] = *reinterpret_cast<const v8h*>(p);       // K: kbase+0..7
            a[mi].h[1] = *reinterpret_cast<const v8h*>(p + 16);  // K: 16+kbase+0..7
        }
        v16h b0 = *reinterpret_cast<const v16h*>(pb);
        v16h b1 = *reinterpret_cast<const v16h*>(pb + 512);
#pragma unroll
        for (int mi = 0; mi < 4; ++mi) {
            acc[2 * mi]     = wmma_f16(a[mi].v, b0, acc[2 * mi]);
            acc[2 * mi + 1] = wmma_f16(a[mi].v, b1, acc[2 * mi + 1]);
        }
        pa += 32;
        pb += bStep;
    }

    // epilogue: per C/D layout lane holds column n, rows mbase..mbase+7
    const int n0 = (snt << 5) + lr;
    float al[2], be[2];
#pragma unroll
    for (int ni = 0; ni < 2; ++ni) {
        const int n = n0 + ni * 16;
        al[ni] = alpha ? alpha[n] : 1.0f;
        be[ni] = beta  ? beta[n]  : 0.0f;
    }
#pragma unroll
    for (int mi = 0; mi < 4; ++mi) {
        const int mbase = smt * 64 + mi * 16 + half * 8;
#pragma unroll
        for (int ni = 0; ni < 2; ++ni) {
            const int n = n0 + ni * 16;
            const v8f& ac = acc[2 * mi + ni];
#pragma unroll
            for (int r = 0; r < 8; ++r) {
                float val = ac[r] * al[ni] + be[ni];
                if (relu) val = fmaxf(val, 0.0f);
                const size_t o = (size_t)(mbase + r) * ldo + n;
                if (out16) out16[o] = (_Float16)val;
                if (out32) out32[o] = val;
            }
        }
    }
}

// Pack a row-major f32 weight [K,N] into WMMA B-fragment order (f16), zero-pad K->Kpad.
__global__ void pack_weights_kernel(const float* __restrict__ W, _Float16* __restrict__ out,
                                    int K, int N, int Kpad)
{
    int idx = blockIdx.x * blockDim.x + threadIdx.x;
    int total = Kpad * N;
    if (idx >= total) return;
    int i    = idx & 15;
    int l    = (idx >> 4) & 31;
    int tile = idx >> 9;
    int NT   = N >> 4;
    int kt   = tile / NT;
    int nt   = tile - kt * NT;
    int k    = kt * 32 + ((i < 8) ? ((l >> 4) * 8 + i) : (16 + (l >> 4) * 8 + (i - 8)));
    int n    = nt * 16 + (l & 15);
    float v  = (k < K) ? W[(size_t)k * N + n] : 0.0f;
    out[idx] = (_Float16)v;
}

// Fold BN (+optional bias) into per-channel alpha/beta.
__global__ void affine_kernel(const float* __restrict__ g, const float* __restrict__ b,
                              const float* __restrict__ m, const float* __restrict__ v,
                              const float* __restrict__ bias,
                              float* __restrict__ alpha, float* __restrict__ beta, int N)
{
    int n = blockIdx.x * blockDim.x + threadIdx.x;
    if (n >= N) return;
    float s = g[n] * rsqrtf(v[n] + 1e-5f);
    alpha[n] = s;
    float bb = b[n] - m[n] * s;
    if (bias) bb += bias[n] * s;
    beta[n] = bb;
}

// Input BN: x[16384,16] -> fts32 (stride 16) and zero-padded f16 copy (stride 32).
__global__ void bn_input_kernel(const float* __restrict__ x,
                                const float* __restrict__ g, const float* __restrict__ b,
                                const float* __restrict__ m, const float* __restrict__ v,
                                float* __restrict__ fts32, _Float16* __restrict__ fts16)
{
    int t = blockIdx.x * blockDim.x + threadIdx.x;
    if (t >= NPTS * 32) return;
    int c = t & 31, i = t >> 5;
    if (c < 16) {
        float s   = g[c] * rsqrtf(v[c] + 1e-5f);
        float val = (x[i * 16 + c] - m[c]) * s + b[c];
        fts32[i * 16 + c] = val;
        fts16[i * 32 + c] = (_Float16)val;
    } else {
        fts16[i * 32 + c] = (_Float16)0.0f;
    }
}

// kNN (k=16, self excluded): one block (128 thr) per jet, features staged in LDS.
__global__ void knn_kernel(const float* __restrict__ fts, int C, int* __restrict__ out)
{
    extern __shared__ float sf[];             // 128*C floats
    int bjet = blockIdx.x, p = threadIdx.x;
    for (int t = p; t < PJET * C; t += PJET)
        sf[t] = fts[(size_t)bjet * PJET * C + t];
    __syncthreads();
    float dist[KNN]; int nid[KNN];
#pragma unroll
    for (int j = 0; j < KNN; ++j) { dist[j] = 3.0e38f; nid[j] = 0; }
    const float* fp = sf + p * C;
    for (int q = 0; q < PJET; ++q) {
        if (q == p) continue;
        const float* fq = sf + q * C;
        float d = 0.0f;
        for (int c = 0; c < C; ++c) { float df = fp[c] - fq[c]; d += df * df; }
        if (d < dist[KNN - 1]) {
            float dc = d; int ic = q;
#pragma unroll
            for (int j = 0; j < KNN; ++j) {
                if (dc < dist[j]) {
                    float td = dist[j]; int ti = nid[j];
                    dist[j] = dc; nid[j] = ic; dc = td; ic = ti;
                }
            }
        }
    }
    int gi = bjet * PJET + p;
#pragma unroll
    for (int j = 0; j < KNN; ++j) out[gi * KNN + j] = nid[j];
}

// Build edge features [x_i, x_j - x_i] (f16) for one 16-jet chunk.
__global__ void gather_kernel(const float* __restrict__ fts, const int* __restrict__ knn,
                              _Float16* __restrict__ out, int C, int b0)
{
    int C2 = C * 2;
    long t = (long)blockIdx.x * blockDim.x + threadIdx.x;
    long total = (long)CHEDGE * C2;
    if (t >= total) return;
    int c2 = (int)(t % C2);
    int e  = (int)(t / C2);
    int j  = e & 15;
    int pt = e >> 4;                          // local point 0..2047
    int gi = b0 * PJET + pt;                  // global point index
    int qn = knn[gi * KNN + j];
    int bl = pt >> 7;                         // local jet
    int gq = (b0 + bl) * PJET + qn;
    int c  = (c2 < C) ? c2 : (c2 - C);
    float xi  = fts[(size_t)gi * C + c];
    float val = (c2 < C) ? xi : (fts[(size_t)gq * C + c] - xi);
    out[t] = (_Float16)val;
}

// Mean over k=16 neighbors for one chunk.
__global__ void agg_kernel(const _Float16* __restrict__ h, float* __restrict__ agg,
                           int Cout, int b0)
{
    int t = blockIdx.x * blockDim.x + threadIdx.x;
    if (t >= CHPTS * Cout) return;
    int c  = t % Cout;
    int pt = t / Cout;
    const _Float16* p = h + (size_t)(pt * KNN) * Cout + c;
    float s = 0.0f;
#pragma unroll
    for (int j = 0; j < KNN; ++j) s += (float)p[j * Cout];
    agg[(size_t)(b0 * PJET + pt) * Cout + c] = s * (1.0f / 16.0f);
}

// relu(agg + shortcut) -> new fts (f32 + f16) and slot into fusion concat buffer.
__global__ void combine_kernel(const float* __restrict__ agg, const float* __restrict__ sc,
                               float* __restrict__ fts32, _Float16* __restrict__ fts16,
                               _Float16* __restrict__ cat, int Cout, int coff)
{
    int t = blockIdx.x * blockDim.x + threadIdx.x;
    if (t >= NPTS * Cout) return;
    int c = t % Cout, i = t / Cout;
    float v = fmaxf(agg[t] + sc[t], 0.0f);
    fts32[t] = v;
    fts16[t] = (_Float16)v;
    cat[(size_t)i * 448 + coff + c] = (_Float16)v;
}

// Global mean pool over P -> f16 [128,384] for fc1 GEMM.
__global__ void pool_kernel(const float* __restrict__ fused, _Float16* __restrict__ p16)
{
    int t = blockIdx.x * blockDim.x + threadIdx.x;
    if (t >= 128 * 384) return;
    int c = t % 384, b = t / 384;
    const float* f = fused + (size_t)b * PJET * 384 + c;
    float s = 0.0f;
    for (int p = 0; p < PJET; ++p) s += f[p * 384];
    p16[t] = (_Float16)(s * (1.0f / 128.0f));
}

// fc2: [128,256] x [256,5] + b (N=5 not WMMA-friendly -> scalar).
__global__ void fc2_kernel(const float* __restrict__ h1, const float* __restrict__ W,
                           const float* __restrict__ b, float* __restrict__ out)
{
    int t = blockIdx.x * blockDim.x + threadIdx.x;
    if (t >= 128 * 5) return;
    int n = t % 5, bb = t / 5;
    float s = b[n];
    for (int k = 0; k < 256; ++k) s += h1[bb * 256 + k] * W[k * 5 + n];
    out[t] = s;
}

// ---------------------------------------------------------------------------
// Host orchestration. Input flattening (setup_inputs dict insertion order):
// 0:x 1:batch 2-5:bn_fts(g,b,m,v)
// conv blocks (23 entries each) at 6/29/52:
//   per mlp layer l: +6l:W +6l+1:b +6l+2..5:bn(g,b,m,v); +18:sc_W +19..22:sc_bn
// 75:fusion.W 76-79:fusion.bn 80:fc1_W 81:fc1_b 82:fc2_W 83:fc2_b
// ---------------------------------------------------------------------------
extern "C" void kernel_launch(void* const* d_in, const int* in_sizes, int n_in,
                              void* d_out, int out_size, void* d_ws, size_t ws_size,
                              hipStream_t stream)
{
    (void)in_sizes; (void)n_in; (void)out_size; (void)ws_size;
    auto F = [&](int i) -> const float* { return (const float*)d_in[i]; };

    size_t off = 0;
    auto alloc = [&](size_t bytes) -> char* {
        char* p = (char*)d_ws + off;
        off += (bytes + 255) & ~(size_t)255;
        return p;
    };
    float*    fts32    = (float*)   alloc((size_t)NPTS * 256 * 4);
    _Float16* fts16    = (_Float16*)alloc((size_t)NPTS * 256 * 2);
    int*      knnIdx   = (int*)     alloc((size_t)NPTS * KNN * 4);
    _Float16* edgeA    = (_Float16*)alloc((size_t)CHEDGE * 256 * 2);
    _Float16* edgeB    = (_Float16*)alloc((size_t)CHEDGE * 256 * 2);
    float*    aggB     = (float*)   alloc((size_t)NPTS * 256 * 4);
    float*    scb      = (float*)   alloc((size_t)NPTS * 256 * 4);
    _Float16* catB     = (_Float16*)alloc((size_t)NPTS * 448 * 2);
    float*    fused    = (float*)   alloc((size_t)NPTS * 384 * 4);
    _Float16* pooled16 = (_Float16*)alloc((size_t)128 * 384 * 2);
    float*    h1       = (float*)   alloc((size_t)128 * 256 * 4);
    _Float16* wp       = (_Float16*)alloc((size_t)2 * 1024 * 1024);
    float*    afp      = (float*)   alloc((size_t)64 * 1024);

    auto pack = [&](const float* W, int K, int N, int Kpad) -> const _Float16* {
        _Float16* dst = wp; wp += (size_t)Kpad * N;
        int total = Kpad * N;
        pack_weights_kernel<<<(total + 255) / 256, 256, 0, stream>>>(W, dst, K, N, Kpad);
        return dst;
    };
    auto mkaff = [&](const float* g, const float* b, const float* m, const float* v,
                     const float* bias, int N) -> const float* {
        float* a = afp; afp += 2 * N;
        affine_kernel<<<(N + 63) / 64, 64, 0, stream>>>(g, b, m, v, bias, a, a + N, N);
        return a;
    };
    auto gemm = [&](const _Float16* A, int lda, const _Float16* Bp,
                    _Float16* o16, float* o32, int ldo,
                    const float* alpha, const float* beta,
                    int M, int N, int K, int relu) {
        int stiles = (M / 64) * (N / 32);     // 4x2 super-tiles per wave
        gemm_wmma_kernel<<<(stiles + 7) / 8, 256, 0, stream>>>(
            A, lda, Bp, o16, o32, ldo, alpha, beta, M, N, K, relu);
    };

    // input BN
    bn_input_kernel<<<(NPTS * 32 + 255) / 256, 256, 0, stream>>>(
        F(0), F(2), F(3), F(4), F(5), fts32, fts16);

    const int convBase[3] = {6, 29, 52};
    const int convCin[3]  = {16, 64, 128};
    const int convCout[3] = {64, 128, 256};
    int coff = 0;
    for (int ci = 0; ci < 3; ++ci) {
        const int base = convBase[ci], Cin = convCin[ci], Cout = convCout[ci];
        const int din = 2 * Cin;
        const int KpadIn = (Cin < 32) ? 32 : Cin;

        const _Float16* pk[3];
        const float*    af[3];
        for (int l = 0; l < 3; ++l) {
            int K = (l == 0) ? din : Cout;
            pk[l] = pack(F(base + 6 * l), K, Cout, K);
            af[l] = mkaff(F(base + 6 * l + 2), F(base + 6 * l + 3),
                          F(base + 6 * l + 4), F(base + 6 * l + 5),
                          F(base + 6 * l + 1), Cout);
        }
        const _Float16* pkSc = pack(F(base + 18), Cin, Cout, KpadIn);
        const float*    afSc = mkaff(F(base + 19), F(base + 20), F(base + 21),
                                     F(base + 22), nullptr, Cout);

        knn_kernel<<<128, 128, (size_t)PJET * Cin * 4, stream>>>(fts32, Cin, knnIdx);

        for (int ch = 0; ch < 8; ++ch) {
            const int b0 = ch * CBJ;
            long tg = (long)CHEDGE * din;
            gather_kernel<<<(int)((tg + 255) / 256), 256, 0, stream>>>(
                fts32, knnIdx, edgeA, Cin, b0);
            gemm(edgeA, din,  pk[0], edgeB, nullptr, Cout, af[0], af[0] + Cout,
                 CHEDGE, Cout, din,  1);
            gemm(edgeB, Cout, pk[1], edgeA, nullptr, Cout, af[1], af[1] + Cout,
                 CHEDGE, Cout, Cout, 1);
            gemm(edgeA, Cout, pk[2], edgeB, nullptr, Cout, af[2], af[2] + Cout,
                 CHEDGE, Cout, Cout, 1);
            int ta = CHPTS * Cout;
            agg_kernel<<<(ta + 255) / 256, 256, 0, stream>>>(edgeB, aggB, Cout, b0);
        }
        // shortcut GEMM over all points (reads current fts16 before combine overwrites it)
        gemm(fts16, KpadIn, pkSc, nullptr, scb, Cout, afSc, afSc + Cout,
             NPTS, Cout, KpadIn, 0);
        int tc = NPTS * Cout;
        combine_kernel<<<(tc + 255) / 256, 256, 0, stream>>>(
            aggB, scb, fts32, fts16, catB, Cout, coff);
        coff += Cout;
    }

    // fusion 448 -> 384 (BN + relu), pool, fc1 (WMMA), fc2 (scalar)
    const _Float16* pkF = pack(F(75), 448, 384, 448);
    const float*    afF = mkaff(F(76), F(77), F(78), F(79), nullptr, 384);
    gemm(catB, 448, pkF, nullptr, fused, 384, afF, afF + 384, NPTS, 384, 448, 1);

    pool_kernel<<<(128 * 384 + 255) / 256, 256, 0, stream>>>(fused, pooled16);

    const _Float16* pkFc1 = pack(F(80), 384, 256, 384);
    gemm(pooled16, 384, pkFc1, nullptr, h1, 256, nullptr, F(81), 128, 256, 384, 1);

    fc2_kernel<<<(640 + 255) / 256, 256, 0, stream>>>(h1, F(82), F(83), (float*)d_out);
}
